// Nemo_88545045774373
// MI455X (gfx1250) — matrix operand order repaired and checked by reading, not compiled, not run on hardware
//
#include <hip/hip_runtime.h>
#include <hip/hip_bf16.h>

// ---------------------------------------------------------------------------
// Fused hashgrid-encode + 3-layer MLP for MI455X (gfx1250, wave32, WMMA).
//   enc:  [N,64]  f16 (built in LDS from dense multires grid gathers)
//   h1 = relu(enc @ W1 + b1)   v_wmma_f32_16x16x32_f16, K=64  (B-stationary)
//   h2 = relu(h1  @ W2 + b2)   v_wmma_f32_16x16x32_f16, K=256 (B-stationary)
//   z  = h2 @ W3 + b3          folded into layer-2 epilogue (shfl + ds_add_f32)
// All tile loops have compile-time trip counts so EXEC stays all-ones
// around every WMMA (ISA requirement). A-fragments for all 4 row tiles are
// loaded into distinct registers before the 4 WMMAs so the ds_loads clause
// behind a single s_wait_dscnt and the WMMAs issue back-to-back.
// ---------------------------------------------------------------------------

typedef __attribute__((ext_vector_type(16))) _Float16 v16h;
typedef __attribute__((ext_vector_type(8)))  _Float16 v8h;
typedef __attribute__((ext_vector_type(8)))  float    v8f;
typedef __attribute__((ext_vector_type(2)))  int      v2i;

#define LVLS 8
#define FPL  8
#define TSZ  (1 << 19)
#define MBLK 64            // points per block
#define ENC_S 72           // enc LDS row stride (halves): 144B = 9*16B -> conflict-free b128
#define H1_S  264          // h1 LDS row stride (halves): 528B = 33*16B -> conflict-free b128

#if __has_builtin(__builtin_amdgcn_global_load_async_to_lds_b64) && \
    __has_builtin(__builtin_amdgcn_s_wait_asynccnt)
#define HAVE_ASYNC_LDS 1
typedef __attribute__((address_space(1))) v2i* as1_v2i_p;   // "v2i __device__*"
typedef __attribute__((address_space(3))) v2i* as3_v2i_p;   // "v2i __shared__*"
#else
#define HAVE_ASYNC_LDS 0
#endif

__constant__ float c_scale[LVLS] = {
    15.0f, 19.158737f, 24.398417f, 31.000002f,
    39.317474f, 49.796833f, 63.000001f, 79.634949f};
__constant__ int c_res[LVLS] = {16, 21, 26, 32, 40, 51, 64, 81};

// Convert W1[64,256], W2[256,256] (row-major f32, [k][n]) into f16 column-major
// [n][K] so WMMA B-fragments are contiguous-per-lane.
__global__ void Nemo_convert_w(const float* __restrict__ W1,
                               const float* __restrict__ W2,
                               _Float16* __restrict__ Wt1,
                               _Float16* __restrict__ Wt2) {
    int tid = blockIdx.x * blockDim.x + threadIdx.x;
    if (tid < 64 * 256) {
        int n = tid & 255, k = tid >> 8;              // k in [0,64)
        Wt1[n * 64 + k] = (_Float16)W1[k * 256 + n];
    }
    int t2 = tid - 64 * 256;
    if (t2 >= 0 && t2 < 256 * 256) {
        int n = t2 & 255, k = t2 >> 8;
        Wt2[n * 256 + k] = (_Float16)W2[k * 256 + n];
    }
}

// A-fragment (16x32 f16) from row-major LDS tile, per CDNA5 WMMA VGPR layout:
// lanes 0-15: row m0+lane, K = k0+[0..7] and k0+16+[0..7]
// lanes 16-31: row m0+lane-16, K = k0+8+[0..7] and k0+24+[0..7]
__device__ __forceinline__ v16h load_a_lds(const _Float16* base, int stride,
                                           int m0, int k0, int lane) {
    int row = m0 + (lane & 15);
    int kh  = (lane < 16) ? 0 : 8;
    union { v16h v; v8h h[2]; } u;
    u.h[0] = *(const v8h*)(base + row * stride + k0 + kh);
    u.h[1] = *(const v8h*)(base + row * stride + k0 + 16 + kh);
    return u.v;
}

// B-fragment (32x16 f16) from column-major global weights Wt[n][KDIM]:
// lanes 0-15: col n0+lane, K = k0+[0..7] and k0+16+[0..7]; lanes 16-31: col n0+lane-16, K = k0+8+[0..7] and k0+24+[0..7]
__device__ __forceinline__ v16h load_b_glb(const _Float16* __restrict__ Wt, int KDIM,
                                           int n0, int k0, int lane) {
    int col = n0 + (lane & 15);
    int kb  = k0 + ((lane < 16) ? 0 : 8);
    const _Float16* p = Wt + col * KDIM + kb;
    union { v16h v; v8h h[2]; } u;
    u.h[0] = *(const v8h*)(p);
    u.h[1] = *(const v8h*)(p + 16);
    return u.v;
}

__global__ void __launch_bounds__(256)
Nemo_fused_mlp(const float* __restrict__ x,
               const float* __restrict__ tables,
               const _Float16* __restrict__ Wt1,
               const _Float16* __restrict__ Wt2,
               const float* __restrict__ b1,
               const float* __restrict__ b2,
               const float* __restrict__ W3,
               const float* __restrict__ b3,
               float* __restrict__ out) {
    __shared__ _Float16 sEnc[MBLK * ENC_S];   // [64][72] halves
    __shared__ _Float16 sH1[MBLK * H1_S];     // [64][264] halves
    __shared__ float    sX[MBLK * 2];         // staged xy pairs
    __shared__ float    sOut[MBLK];

    const int tid  = threadIdx.x;
    const int lane = tid & 31;
    const int wave = tid >> 5;

    // Warm L2/L1 with the weight images (global_prefetch_b8 path).
    __builtin_prefetch(Wt1 + lane * 256, 0, 1);
    __builtin_prefetch(Wt2 + lane * 2048, 0, 1);

    // Stage this block's 64 xy pairs into LDS (async-to-LDS path when present).
    if (tid < MBLK) {
        sOut[tid] = 0.0f;
        const float* gx = x + ((size_t)blockIdx.x * MBLK + tid) * 2;
#if HAVE_ASYNC_LDS
        __builtin_amdgcn_global_load_async_to_lds_b64(
            (as1_v2i_p)gx, (as3_v2i_p)(&sX[tid * 2]), 0, 0);
#else
        sX[tid * 2 + 0] = gx[0];
        sX[tid * 2 + 1] = gx[1];
#endif
    }
#if HAVE_ASYNC_LDS
    __builtin_amdgcn_s_wait_asynccnt(0);
#endif
    __syncthreads();

    // ---- Encode: 4 threads per point, 2 levels each -> f16 features in LDS ----
    {
        const int p = tid >> 2;               // 0..63
        const int q = tid & 3;                // level pair id
        const float px = sX[p * 2 + 0];
        const float py = sX[p * 2 + 1];
        #pragma unroll
        for (int li = 0; li < 2; ++li) {
            const int l = q * 2 + li;
            const float scale = c_scale[l];
            const int   res   = c_res[l];
            const float fx = px * scale + 0.5f;
            const float fy = py * scale + 0.5f;
            const float x0f = floorf(fx), y0f = floorf(fy);
            const float tx = fx - x0f, ty = fy - y0f;
            const float wx = tx * tx * (3.0f - 2.0f * tx);   // smoothstep
            const float wy = ty * ty * (3.0f - 2.0f * ty);
            const int x0 = (int)x0f, y0 = (int)y0f;
            const float* tb = tables + (size_t)l * TSZ * FPL;
            float acc[FPL];
            #pragma unroll
            for (int f = 0; f < FPL; ++f) acc[f] = 0.0f;
            #pragma unroll
            for (int cy = 0; cy < 2; ++cy) {
                #pragma unroll
                for (int cx = 0; cx < 2; ++cx) {
                    const int idx = (x0 + cx) + (y0 + cy) * res;   // dense stride < TSZ
                    const float w = (cx ? wx : 1.0f - wx) * (cy ? wy : 1.0f - wy);
                    const float4* src = (const float4*)(tb + (size_t)idx * FPL);
                    const float4 a = src[0], b = src[1];
                    acc[0] += a.x * w; acc[1] += a.y * w;
                    acc[2] += a.z * w; acc[3] += a.w * w;
                    acc[4] += b.x * w; acc[5] += b.y * w;
                    acc[6] += b.z * w; acc[7] += b.w * w;
                }
            }
            v8h hv;
            #pragma unroll
            for (int f = 0; f < FPL; ++f) hv[f] = (_Float16)acc[f];
            *(v8h*)(&sEnc[p * ENC_S + l * FPL]) = hv;   // ds_store_b128 (16B aligned)
        }
    }
    __syncthreads();

    // ---- Layer 1: B-stationary. Wave owns column panels nt = wave, wave+8. ----
    #pragma unroll
    for (int i = 0; i < 2; ++i) {
        const int n0 = (wave + 8 * i) * 16;
        const float bv = b1[n0 + (lane & 15)];
        v8f acc[4];
        #pragma unroll
        for (int mt = 0; mt < 4; ++mt) {
            v8f c = {bv, bv, bv, bv, bv, bv, bv, bv};
            acc[mt] = c;
        }
        #pragma unroll
        for (int k0 = 0; k0 < 64; k0 += 32) {
            const v16h b = load_b_glb(Wt1, 64, n0, k0, lane);   // once per k-step
            // distinct locals -> clause the 8 ds_loads, one wait, 4 WMMAs b2b
            const v16h a0 = load_a_lds(sEnc, ENC_S,  0, k0, lane);
            const v16h a1 = load_a_lds(sEnc, ENC_S, 16, k0, lane);
            const v16h a2 = load_a_lds(sEnc, ENC_S, 32, k0, lane);
            const v16h a3 = load_a_lds(sEnc, ENC_S, 48, k0, lane);
            acc[0] = __builtin_amdgcn_wmma_f32_16x16x32_f16(false, a0, false, b, (short)0, acc[0], false, false);
            acc[1] = __builtin_amdgcn_wmma_f32_16x16x32_f16(false, a1, false, b, (short)0, acc[1], false, false);
            acc[2] = __builtin_amdgcn_wmma_f32_16x16x32_f16(false, a2, false, b, (short)0, acc[2], false, false);
            acc[3] = __builtin_amdgcn_wmma_f32_16x16x32_f16(false, a3, false, b, (short)0, acc[3], false, false);
        }
        const int colc = n0 + (lane & 15);
        #pragma unroll
        for (int mt = 0; mt < 4; ++mt) {
            const int rbase = mt * 16 + ((lane < 16) ? 0 : 8);
            #pragma unroll
            for (int r = 0; r < 8; ++r) {
                float v = fmaxf(acc[mt][r], 0.0f);
                sH1[(rbase + r) * H1_S + colc] = (_Float16)v;
            }
        }
    }
    __syncthreads();

    // ---- Layer 2 (K=256, B-stationary) fused with layer 3 (dot with W3) ----
    #pragma unroll
    for (int i = 0; i < 2; ++i) {
        const int n0 = (wave + 8 * i) * 16;
        const float bv = b2[n0 + (lane & 15)];
        v8f acc[4];
        #pragma unroll
        for (int mt = 0; mt < 4; ++mt) {
            v8f c = {bv, bv, bv, bv, bv, bv, bv, bv};
            acc[mt] = c;
        }
        #pragma unroll
        for (int k0 = 0; k0 < 256; k0 += 32) {
            const v16h b = load_b_glb(Wt2, 256, n0, k0, lane);  // once per k-step
            const v16h a0 = load_a_lds(sH1, H1_S,  0, k0, lane);
            const v16h a1 = load_a_lds(sH1, H1_S, 16, k0, lane);
            const v16h a2 = load_a_lds(sH1, H1_S, 32, k0, lane);
            const v16h a3 = load_a_lds(sH1, H1_S, 48, k0, lane);
            acc[0] = __builtin_amdgcn_wmma_f32_16x16x32_f16(false, a0, false, b, (short)0, acc[0], false, false);
            acc[1] = __builtin_amdgcn_wmma_f32_16x16x32_f16(false, a1, false, b, (short)0, acc[1], false, false);
            acc[2] = __builtin_amdgcn_wmma_f32_16x16x32_f16(false, a2, false, b, (short)0, acc[2], false, false);
            acc[3] = __builtin_amdgcn_wmma_f32_16x16x32_f16(false, a3, false, b, (short)0, acc[3], false, false);
        }
        const float w3v = W3[n0 + (lane & 15)];
        #pragma unroll
        for (int mt = 0; mt < 4; ++mt) {
            const int rbase = mt * 16 + ((lane < 16) ? 0 : 8);
            #pragma unroll
            for (int r = 0; r < 8; ++r) {
                float v = fmaxf(acc[mt][r], 0.0f) * w3v;
                // reduce the 16 N-columns held by this 16-lane half-group
                v += __shfl_xor(v, 1, 32);
                v += __shfl_xor(v, 2, 32);
                v += __shfl_xor(v, 4, 32);
                v += __shfl_xor(v, 8, 32);
                if ((lane & 15) == 0) atomicAdd(&sOut[rbase + r], v);  // ds_add_f32
            }
        }
    }
    __syncthreads();

    if (tid < MBLK)
        out[(size_t)blockIdx.x * MBLK + tid] = sOut[tid] + b3[0];
}

extern "C" void kernel_launch(void* const* d_in, const int* in_sizes, int n_in,
                              void* d_out, int out_size, void* d_ws, size_t ws_size,
                              hipStream_t stream) {
    const float* x      = (const float*)d_in[0];
    const float* tables = (const float*)d_in[1];
    const float* W1     = (const float*)d_in[2];
    const float* b1     = (const float*)d_in[3];
    const float* W2     = (const float*)d_in[4];
    const float* b2     = (const float*)d_in[5];
    const float* W3     = (const float*)d_in[6];
    const float* b3     = (const float*)d_in[7];
    float* out = (float*)d_out;

    _Float16* Wt1 = (_Float16*)d_ws;           // 64*256 halves = 32 KB
    _Float16* Wt2 = Wt1 + 64 * 256;            // 128 KB

    const int N = in_sizes[0] / 2;             // 524288 points

    Nemo_convert_w<<<(64 * 256 + 256 * 256 + 255) / 256, 256, 0, stream>>>(W1, W2, Wt1, Wt2);
    Nemo_fused_mlp<<<N / MBLK, 256, 0, stream>>>(x, tables, Wt1, Wt2, b1, b2, W3, b3, out);
}
